// AttLoc_47631187313261
// MI455X (gfx1250) — matrix operand order repaired and checked by reading, not compiled, not run on hardware
//
#include <hip/hip_runtime.h>
#include <hip/hip_bf16.h>

typedef __attribute__((ext_vector_type(16))) __bf16 v16bf;
typedef __attribute__((ext_vector_type(8)))  __bf16 v8bf;
typedef __attribute__((ext_vector_type(4)))  __bf16 v4bf;
typedef __attribute__((ext_vector_type(8)))  float  v8f;

constexpr int kB      = 64;
constexpr int kT      = 2048;
constexpr int kE      = 512;    // EPROJS
constexpr int kD      = 1024;   // DUNITS
constexpr int kA      = 320;    // ATT_DIM
constexpr int kC      = 10;     // ACONV_CHANS
constexpr int kF      = 100;    // ACONV_FILTS
constexpr int kKW     = 201;    // 2*filts+1
constexpr int kTT     = 32;     // t rows per tile
constexpr int kNT     = kT / kTT;   // 64 tiles per batch row
constexpr int kLDSK   = kE + 8;     // padded LDS row (bank-conflict-free b128)

// hardware tanh (V_TANH_F32) on gfx1250 if the builtin exists
#if __has_builtin(__builtin_amdgcn_tanhf)
__device__ __forceinline__ float fast_tanh(float x) { return __builtin_amdgcn_tanhf(x); }
#else
__device__ __forceinline__ float fast_tanh(float x) { return tanhf(x); }
#endif

// ---------------- workspace layout (bytes) ----------------
constexpr size_t OFF_CONV = 0;                                  // [B,T,10] f32
constexpr size_t SZ_CONV  = (size_t)kB * kT * kC * 4;
constexpr size_t OFF_BIAS = OFF_CONV + SZ_CONV;                 // [B,320] f32
constexpr size_t SZ_BIAS  = (size_t)kB * kA * 4;
constexpr size_t OFF_WENC = OFF_BIAS + SZ_BIAS;                 // [320,512] bf16
constexpr size_t SZ_WENC  = (size_t)kA * kE * 2;
constexpr size_t OFF_E2   = OFF_WENC + SZ_WENC;                 // [B,T] f32 (scaled, masked)
constexpr size_t SZ_E2    = (size_t)kB * kT * 4;
constexpr size_t OFF_M    = OFF_E2 + SZ_E2;                     // [B,64] tile max
constexpr size_t SZ_MS    = (size_t)kB * kNT * 4;
constexpr size_t OFF_S    = OFF_M + SZ_MS;                      // [B,64] tile sumexp
constexpr size_t OFF_CP   = OFF_S + SZ_MS;                      // [B,64,512] partial ctx

// ============ K1: location-aware 1D conv (SAME, 201 taps, 10 ch) ============
__global__ __launch_bounds__(256) void loc_conv_kernel(
    const float* __restrict__ att_prev, const float* __restrict__ conv_w,
    float* __restrict__ aconv) {
  __shared__ float win[256 + 2 * kF];
  __shared__ float cw[kC * kKW];
  const int b = blockIdx.y, t0 = blockIdx.x * 256, tid = threadIdx.x;
  for (int i = tid; i < 256 + 2 * kF; i += 256) {
    int g = t0 - kF + i;
    win[i] = (g >= 0 && g < kT) ? att_prev[(size_t)b * kT + g] : 0.0f;
  }
  for (int i = tid; i < kC * kKW; i += 256) cw[i] = conv_w[i];
  __syncthreads();
  float s[kC];
#pragma unroll
  for (int c = 0; c < kC; ++c) s[c] = 0.0f;
  for (int j = 0; j < kKW; ++j) {
    const float x = win[tid + j];
#pragma unroll
    for (int c = 0; c < kC; ++c) s[c] += cw[c * kKW + j] * x;
  }
  float* dst = aconv + ((size_t)b * kT + t0 + tid) * kC;
#pragma unroll
  for (int c = 0; c < kC; ++c) dst[c] = s[c];
}

// ============ K2: bias[b,a] = b_enc[a] + dec_z[b,:] . W_dec[a,:] ============
__global__ __launch_bounds__(320) void dec_bias_kernel(
    const float* __restrict__ dec_z, const float* __restrict__ W_dec,
    const float* __restrict__ b_enc, float* __restrict__ bias) {
  const int b = blockIdx.x, a = threadIdx.x;
  const float* z = dec_z + (size_t)b * kD;
  const float* w = W_dec + (size_t)a * kD;
  float s = b_enc[a];
  for (int d = 0; d < kD; ++d) s += z[d] * w[d];
  bias[b * kA + a] = s;
}

// ============ K3: W_enc fp32 -> bf16 (one-time, stays L2 resident) ============
__global__ __launch_bounds__(256) void cvt_wenc_kernel(
    const float* __restrict__ W, __bf16* __restrict__ Wb) {
  const size_t i = (size_t)blockIdx.x * 256 + threadIdx.x;
  Wb[i] = (__bf16)W[i];
}

// ============ K4: fused energies + tile softmax stats + partial context ======
// grid = (kNT tiles, kB); block = 256 threads = 8 waves.
__global__ __launch_bounds__(256) void att_energy_kernel(
    const float* __restrict__ enc, const int* __restrict__ lens,
    const float* __restrict__ Watt, const float* __restrict__ Wg,
    const float* __restrict__ bg, const float* __restrict__ bias,
    const __bf16* __restrict__ wenc, const float* __restrict__ aconv,
    float* __restrict__ e2_out, float* __restrict__ m_out,
    float* __restrict__ s_out, float* __restrict__ c_part) {
  __shared__ __bf16 lds_enc[kTT * kLDSK];   // 33.3 KB, padded rows
  __shared__ float  lds_conv[kTT * kC];
  __shared__ float  lds_pe[8 * 16];          // per-wave energy partials
  __shared__ float  lds_p[kTT];

  const int tile = blockIdx.x, b = blockIdx.y, tid = threadIdx.x;
  const int t0 = tile * kTT;

  // ---- phase 0: stage enc tile (fp32 -> bf16) + conv tile into LDS ----
  {
    const float* src = enc + ((size_t)b * kT + t0) * kE;
#pragma unroll
    for (int i = 0; i < 16; ++i) {
      const int elem = (tid + i * 256) * 4;      // 32*512 fp32 / 256 thr
      const int row = elem >> 9, col = elem & (kE - 1);
      const float4 f = *(const float4*)(src + row * kE + col);
      v4bf v;
      v[0] = (__bf16)f.x; v[1] = (__bf16)f.y;
      v[2] = (__bf16)f.z; v[3] = (__bf16)f.w;
      *(v4bf*)(&lds_enc[row * kLDSK + col]) = v;
    }
    for (int i = tid; i < kTT * kC; i += 256)
      lds_conv[i] = aconv[((size_t)b * kT + t0) * kC + i];
  }
  __syncthreads();

  // ---- phase 1: WMMA GEMM  [32 x 512] x [512 x 320] with fused epilogue ----
  const int wave = tid >> 5, lane = tid & 31;
  const int khalf = lane >> 4, ln = lane & 15;
  const int msub = wave & 1;            // which 16-row M subtile
  const int m0 = msub * 16;
  const int ngrp = wave >> 1;           // 4 wave-groups split 20 N tiles
  const int row_base = m0 + 8 * khalf;  // C-matrix rows held by this lane
  const float bgv = bg[0];

  float pe[8];
#pragma unroll
  for (int r = 0; r < 8; ++r) pe[r] = 0.0f;

  for (int nt = ngrp; nt < kA / 16; nt += 4) {
    const int a0 = nt * 16;
    const int a = a0 + ln;
    // seed accumulator with bias + att_feat  (never materialize [B,T,320])
    v8f acc;
    const float wb = bias[b * kA + a];
    float wat[kC];
#pragma unroll
    for (int c = 0; c < kC; ++c) wat[c] = Watt[a * kC + c];
#pragma unroll
    for (int r = 0; r < 8; ++r) {
      const int row = row_base + r;
      float s = wb;
#pragma unroll
      for (int c = 0; c < kC; ++c) s += lds_conv[row * kC + c] * wat[c];
      acc[r] = s;
    }
    // K loop: 512 = 16 x 32
    const __bf16* brow = wenc + (size_t)a * kE;              // B column a
    const __bf16* arow = lds_enc + (m0 + ln) * kLDSK;        // A row
#pragma unroll 4
    for (int kk = 0; kk < kE / 32; ++kk) {
      const int k0 = kk * 32;
      // A 16x32 bf16 layout: VGPR0-3 = K khalf*8..+7, VGPR4-7 = K 16+khalf*8..+7
      const v8bf alo = *(const v8bf*)(arow + k0 + khalf * 8);
      const v8bf ahi = *(const v8bf*)(arow + k0 + 16 + khalf * 8);
      v16bf af;
#pragma unroll
      for (int i = 0; i < 8; ++i) { af[i] = alo[i]; af[8 + i] = ahi[i]; }
      // B 32x16 bf16 layout: same as A, K khalf*8..+7 then 16+khalf*8..+7
      const v8bf blo = *(const v8bf*)(brow + k0 + khalf * 8);
      const v8bf bhi = *(const v8bf*)(brow + k0 + 16 + khalf * 8);
      const v16bf bfm = __builtin_shufflevector(blo, bhi, 0, 1, 2, 3, 4, 5, 6, 7, 8, 9, 10, 11, 12, 13, 14, 15);
      acc = __builtin_amdgcn_wmma_f32_16x16x32_bf16(
          false, af, false, bfm, (short)0, acc, false, false);
    }
    const float wg = Wg[a];
#pragma unroll
    for (int r = 0; r < 8; ++r) pe[r] += wg * fast_tanh(acc[r]);
  }
  // reduce over 16 N-lanes within each half (rows differ per half)
#pragma unroll
  for (int r = 0; r < 8; ++r) {
    float v = pe[r];
    for (int mask = 1; mask < 16; mask <<= 1) v += __shfl_xor(v, mask, 32);
    pe[r] = v;
  }
  if (ln == 0) {  // lane 0 (rows m0..m0+7) and lane 16 (rows m0+8..m0+15)
#pragma unroll
    for (int r = 0; r < 8; ++r) lds_pe[wave * 16 + 8 * khalf + r] = pe[r];
  }
  __syncthreads();

  // ---- phase 2: mask + scale + tile-local softmax stats (wave 0) ----
  if (tid < kTT) {
    const int row = tid, ms = row >> 4, ri = row & 15;
    float e = bgv;
#pragma unroll
    for (int i = 0; i < 4; ++i) e += lds_pe[(ms + 2 * i) * 16 + ri];
    const int tg = t0 + row;
    const float e2 = (tg < lens[b]) ? 2.0f * e : -1.0e30f;  // SCALING folded
    e2_out[(size_t)b * kT + tg] = e2;
    float m = e2;
    for (int mask = 16; mask >= 1; mask >>= 1)
      m = fmaxf(m, __shfl_xor(m, mask, 32));
    const float p = __expf(e2 - m);
    lds_p[row] = p;
    float ss = p;
    for (int mask = 16; mask >= 1; mask >>= 1) ss += __shfl_xor(ss, mask, 32);
    if (tid == 0) {
      m_out[b * kNT + tile] = m;
      s_out[b * kNT + tile] = ss;
    }
  }
  __syncthreads();

  // ---- phase 3: partial context from the LDS-resident tile ----
#pragma unroll
  for (int cix = 0; cix < 2; ++cix) {
    const int col = tid + cix * 256;
    float accc = 0.0f;
#pragma unroll 8
    for (int r = 0; r < kTT; ++r)
      accc += lds_p[r] * (float)lds_enc[r * kLDSK + col];
    c_part[((size_t)(b * kNT + tile)) * kE + col] = accc;
  }
}

// ============ K5: combine tiles (log-sum-exp), emit c and w ==================
__global__ __launch_bounds__(256) void att_finalize_kernel(
    const float* __restrict__ m_in, const float* __restrict__ s_in,
    const float* __restrict__ c_part, const float* __restrict__ e2_in,
    float* __restrict__ out) {
  __shared__ float coef[kNT];
  __shared__ float shm, shs;
  const int b = blockIdx.x, tid = threadIdx.x;
  if (tid == 0) {
    float m = -3.0e38f;
    for (int i = 0; i < kNT; ++i) m = fmaxf(m, m_in[b * kNT + i]);
    float s = 0.0f;
    for (int i = 0; i < kNT; ++i) {
      const float c = __expf(m_in[b * kNT + i] - m);
      coef[i] = c;
      s += s_in[b * kNT + i] * c;
    }
    shm = m; shs = s;
  }
  __syncthreads();
  const float m = shm, inv_s = 1.0f / shs;
#pragma unroll
  for (int cix = 0; cix < 2; ++cix) {
    const int col = tid + cix * 256;
    float acc = 0.0f;
    for (int i = 0; i < kNT; ++i)
      acc += c_part[((size_t)(b * kNT + i)) * kE + col] * coef[i];
    out[(size_t)b * kE + col] = acc * inv_s;   // c : [B,512]
  }
  float* wout = out + (size_t)kB * kE;         // w : [B,2048]
#pragma unroll
  for (int k = 0; k < 8; ++k) {
    const int t = tid + k * 256;
    wout[(size_t)b * kT + t] = __expf(e2_in[(size_t)b * kT + t] - m) * inv_s;
  }
}

// ============================ launcher ======================================
extern "C" void kernel_launch(void* const* d_in, const int* in_sizes, int n_in,
                              void* d_out, int out_size, void* d_ws,
                              size_t ws_size, hipStream_t stream) {
  const float* enc      = (const float*)d_in[0];
  const int*   lens     = (const int*)  d_in[1];
  const float* dec_z    = (const float*)d_in[2];
  const float* att_prev = (const float*)d_in[3];
  const float* W_enc    = (const float*)d_in[4];
  const float* b_enc    = (const float*)d_in[5];
  const float* W_dec    = (const float*)d_in[6];
  const float* W_att    = (const float*)d_in[7];
  const float* conv_w   = (const float*)d_in[8];
  const float* W_g      = (const float*)d_in[9];
  const float* b_g      = (const float*)d_in[10];

  char* ws = (char*)d_ws;
  float*  aconv  = (float*)(ws + OFF_CONV);
  float*  bias   = (float*)(ws + OFF_BIAS);
  __bf16* wencb  = (__bf16*)(ws + OFF_WENC);
  float*  e2v    = (float*)(ws + OFF_E2);
  float*  mpart  = (float*)(ws + OFF_M);
  float*  spart  = (float*)(ws + OFF_S);
  float*  cpart  = (float*)(ws + OFF_CP);
  float*  out    = (float*)d_out;

  loc_conv_kernel<<<dim3(kT / 256, kB), 256, 0, stream>>>(att_prev, conv_w, aconv);
  dec_bias_kernel<<<kB, kA, 0, stream>>>(dec_z, W_dec, b_enc, bias);
  cvt_wenc_kernel<<<(kA * kE) / 256, 256, 0, stream>>>(W_enc, wencb);
  att_energy_kernel<<<dim3(kNT, kB), 256, 0, stream>>>(
      enc, lens, W_att, W_g, b_g, bias, wencb, aconv, e2v, mpart, spart, cpart);
  att_finalize_kernel<<<kB, 256, 0, stream>>>(mpart, spart, cpart, e2v, out);
}
